// ModDRDSP_89635967468006
// MI455X (gfx1250) — compile-verified, matrix operand order checked by reading, not run on hardware
//
#include <hip/hip_runtime.h>

// ---------------------------------------------------------------------------
// Types for CDNA5 WMMA (wave32, 16x16x32 bf16 -> f32)
// ---------------------------------------------------------------------------
typedef __attribute__((ext_vector_type(16))) __bf16 v16bf;
typedef __attribute__((ext_vector_type(8)))  float  v8f;

union FragU { uint4 u4[2]; v16bf v; };

__device__ inline v16bf frag_from_lds(const __bf16* rowPtr, int khalf) {
    FragU f;
    f.u4[0] = *(reinterpret_cast<const uint4*>(rowPtr + khalf));
    f.u4[1] = *(reinterpret_cast<const uint4*>(rowPtr + 16 + khalf));
    return f.v;
}

// ---------------------------------------------------------------------------
// bf16 tiled GEMM: C[M,N] = act( A[M,Kp] @ W[N,Kp]^T + bias )
// A, W are bf16 with rows padded (zero-filled) to Kp = multiple of 32.
// 256 threads = 8 waves in 4x2; block tile 128x64, K-step 32.
// Each wave owns a 32x32 output tile: 4 WMMAs per K-step.
// Staging: 3 global b128 loads + 3 LDS b128 stores per thread per step.
// Row/col indices CLAMPED (out-of-range rows feed only never-stored outputs).
// ---------------------------------------------------------------------------
#define GBM 128
#define GBN 64
#define GBK 32
#define GLDT 40  // padded LDS row stride (elements); 80B keeps 16B alignment

__global__ __launch_bounds__(256, 2)
void gemm_wmma_kernel(const __bf16* __restrict__ A, int lda,
                      const __bf16* __restrict__ W, int ldw,
                      const float* __restrict__ bias,
                      float* __restrict__ C, int ldc,
                      int M, int N, int Kp, int act)
{
    __shared__ __align__(16) __bf16 As[GBM * GLDT];
    __shared__ __align__(16) __bf16 Bs[GBN * GLDT];

    const int tid  = threadIdx.x;
    const int lane = tid & 31;
    const int wave = tid >> 5;    // 0..7
    const int wr   = wave >> 1;   // 0..3 : 32-row group
    const int wc   = wave & 1;    // 0..1 : 32-col group
    const int m0   = (int)blockIdx.y * GBM;
    const int n0   = (int)blockIdx.x * GBN;

    v8f acc00 = {0.f,0.f,0.f,0.f,0.f,0.f,0.f,0.f};
    v8f acc01 = acc00, acc10 = acc00, acc11 = acc00;

    const int lrow = tid >> 2;       // 0..63
    const int lk   = (tid & 3) * 8;  // 0,8,16,24  (8 bf16 = 16B per thread)

    // clamped row base pointers (always in-bounds)
    const int ga0 = min(m0 + lrow,      M - 1);
    const int ga1 = min(m0 + lrow + 64, M - 1);
    const int gb  = min(n0 + lrow,      N - 1);
    const __bf16* __restrict__ pa0 = A + (size_t)ga0 * lda + lk;
    const __bf16* __restrict__ pa1 = A + (size_t)ga1 * lda + lk;
    const __bf16* __restrict__ pb  = W + (size_t)gb  * ldw + lk;

    const int r  = lane & 15;
    const int kh = (lane >> 4) * 8;  // lanes 0-15: K 0-7/16-23; 16-31: 8-15/24-31
    const __bf16* fa0 = &As[(wr * 32 + r) * GLDT];
    const __bf16* fa1 = &As[(wr * 32 + 16 + r) * GLDT];
    const __bf16* fb0 = &Bs[(wc * 32 + r) * GLDT];
    const __bf16* fb1 = &Bs[(wc * 32 + 16 + r) * GLDT];

    for (int kb = 0; kb < Kp; kb += GBK) {
        // ---- stage: 3 vector loads, 3 vector LDS stores ----
        const uint4 qa0 = *reinterpret_cast<const uint4*>(pa0 + kb);
        const uint4 qa1 = *reinterpret_cast<const uint4*>(pa1 + kb);
        const uint4 qb  = *reinterpret_cast<const uint4*>(pb  + kb);
        __syncthreads();
        *reinterpret_cast<uint4*>(&As[lrow        * GLDT + lk]) = qa0;
        *reinterpret_cast<uint4*>(&As[(lrow + 64) * GLDT + lk]) = qa1;
        *reinterpret_cast<uint4*>(&Bs[lrow        * GLDT + lk]) = qb;
        __syncthreads();

        // ---- math: 8 ds_load_b128 feed 4 WMMAs ----
        v16bf aF0 = frag_from_lds(fa0, kh);
        v16bf aF1 = frag_from_lds(fa1, kh);
        v16bf bF0 = frag_from_lds(fb0, kh);
        v16bf bF1 = frag_from_lds(fb1, kh);
        acc00 = __builtin_amdgcn_wmma_f32_16x16x32_bf16(false, aF0, false, bF0,
                                                        (short)0, acc00, false, false);
        acc01 = __builtin_amdgcn_wmma_f32_16x16x32_bf16(false, aF0, false, bF1,
                                                        (short)0, acc01, false, false);
        acc10 = __builtin_amdgcn_wmma_f32_16x16x32_bf16(false, aF1, false, bF0,
                                                        (short)0, acc10, false, false);
        acc11 = __builtin_amdgcn_wmma_f32_16x16x32_bf16(false, aF1, false, bF1,
                                                        (short)0, acc11, false, false);
    }

    // ---- epilogue: C/D layout — VGPR r: lanes 0-15 -> M=r, 16-31 -> M=8+r ----
    const int mhi = (lane >> 4) * 8;
    const int cn0 = n0 + wc * 32 + (lane & 15);
    const int cn1 = cn0 + 16;
    const float bv0 = (bias && cn0 < N) ? bias[cn0] : 0.0f;
    const float bv1 = (bias && cn1 < N) ? bias[cn1] : 0.0f;
#pragma unroll
    for (int rr = 0; rr < 8; ++rr) {
        const int m  = m0 + wr * 32 + mhi + rr;
        const int m2 = m + 16;
        if (m < M) {
            if (cn0 < N) {
                float v = acc00[rr] + bv0;
                if (act) v = fmaxf(v, 0.0f);
                C[(size_t)m * ldc + cn0] = v;
            }
            if (cn1 < N) {
                float v = acc01[rr] + bv1;
                if (act) v = fmaxf(v, 0.0f);
                C[(size_t)m * ldc + cn1] = v;
            }
        }
        if (m2 < M) {
            if (cn0 < N) {
                float v = acc10[rr] + bv0;
                if (act) v = fmaxf(v, 0.0f);
                C[(size_t)m2 * ldc + cn0] = v;
            }
            if (cn1 < N) {
                float v = acc11[rr] + bv1;
                if (act) v = fmaxf(v, 0.0f);
                C[(size_t)m2 * ldc + cn1] = v;
            }
        }
    }
}

// ---------------------------------------------------------------------------
// f32 -> bf16 convert with zero-padded rows: out[rows, Kp], in[rows, K]
// ---------------------------------------------------------------------------
__global__ void k_cvt_pad(const float* __restrict__ in, __bf16* __restrict__ out,
                          int rows, int K, int Kp)
{
    int idx = blockIdx.x * blockDim.x + threadIdx.x;
    if (idx >= rows * Kp) return;
    int i = idx / Kp, c = idx % Kp;
    out[idx] = (c < K) ? (__bf16)in[(size_t)i * K + c] : (__bf16)0.0f;
}

// ---------------------------------------------------------------------------
// Fused conv1d(k=8, VALID) + relu + maxpool3(stride 3, VALID)
// ---------------------------------------------------------------------------
__global__ void conv_relu_pool_kernel(const float* __restrict__ x,
                                      const float* __restrict__ w,
                                      const float* __restrict__ b,
                                      float* __restrict__ y,
                                      int Bn, int Cin, int Lin, int Cout, int Lp)
{
    int idx = blockIdx.x * blockDim.x + threadIdx.x;
    int total = Bn * Cout * Lp;
    if (idx >= total) return;
    int p  = idx % Lp;
    int co = (idx / Lp) % Cout;
    int bb = idx / (Lp * Cout);
    const float* wr = w + (size_t)co * Cin * 8;
    const float bias = b[co];
    float m = -3.4e38f;
    for (int j = 0; j < 3; ++j) {
        const int pos = 3 * p + j;
        float s = bias;
        for (int ci = 0; ci < Cin; ++ci) {
            const float* xr = x + ((size_t)bb * Cin + ci) * Lin + pos;
            const float* wk = wr + ci * 8;
#pragma unroll
            for (int k = 0; k < 8; ++k) s += xr[k] * wk[k];
        }
        s = fmaxf(s, 0.0f);
        m = fmaxf(m, s);
    }
    y[idx] = m;
}

// ---------------------------------------------------------------------------
// Graph / GRU elementwise kernels   (HID=300, HID2=600, T=48)
// ---------------------------------------------------------------------------
__global__ void k_atom_message(const float* __restrict__ ia, const float* __restrict__ ib,
                               const int* __restrict__ a2a, const int* __restrict__ a2b,
                               float* __restrict__ out, int rows)
{
    int idx = blockIdx.x * blockDim.x + threadIdx.x;
    if (idx >= rows * 300) return;
    int i = idx / 300, h = idx % 300;
    float s = 0.0f;
#pragma unroll
    for (int j = 0; j < 6; ++j) {
        int pa = a2a[i * 6 + j];
        if (pa != 0) s += ia[(size_t)pa * 300 + h];
        int pb = a2b[i * 6 + j];
        if (pb != 0) s += ib[(size_t)pb * 300 + h];
    }
    out[idx] = s;
}

__global__ void k_rev(const float* __restrict__ ib, const int* __restrict__ b2a,
                      const int* __restrict__ b2revb, float* __restrict__ out, int rows)
{
    int idx = blockIdx.x * blockDim.x + threadIdx.x;
    if (idx >= rows * 300) return;
    int i = idx / 300, h = idx % 300;
    int r = b2revb[i];
    out[idx] = ib[(size_t)r * 300 + h] + ib[(size_t)b2a[r] * 300 + h];
}

__global__ void k_agg(const float* __restrict__ mb, const int* __restrict__ a2b,
                      float* __restrict__ out, int rows)
{
    int idx = blockIdx.x * blockDim.x + threadIdx.x;
    if (idx >= rows * 300) return;
    int i = idx / 300, h = idx % 300;
    float s = 0.0f, mx = -3.4e38f;
#pragma unroll
    for (int j = 0; j < 6; ++j) {
        int p = a2b[i * 6 + j];
        float v = (p != 0) ? mb[(size_t)p * 300 + h] : 0.0f;
        s += v;
        mx = fmaxf(mx, v);
    }
    out[idx] = s * mx;
}

__global__ void k_add(const float* __restrict__ a, const float* __restrict__ b,
                      float* __restrict__ c, int n)
{
    int idx = blockIdx.x * blockDim.x + threadIdx.x;
    if (idx < n) c[idx] = a[idx] + b[idx];
}

// tmp16[rows, 320] = bf16( message_atom[b2a[i]] - rev[i] ), zero pad cols 300..319
__global__ void k_tmp(const float* __restrict__ ma, const float* __restrict__ rev,
                      const int* __restrict__ b2a, __bf16* __restrict__ out, int rows)
{
    int idx = blockIdx.x * blockDim.x + threadIdx.x;
    if (idx >= rows * 320) return;
    int i = idx / 320, c = idx % 320;
    float v = 0.0f;
    if (c < 300) v = ma[(size_t)b2a[i] * 300 + c] - rev[(size_t)i * 300 + c];
    out[idx] = (__bf16)v;
}

// out16[rows,448] = bf16[ atom(119) | masked-gather-sum(src, idxt)(300) | 0-pad ]
__global__ void k_gather_cat(const float* __restrict__ atom, const float* __restrict__ src,
                             const int* __restrict__ idxt, __bf16* __restrict__ out, int rows)
{
    int idx = blockIdx.x * blockDim.x + threadIdx.x;
    if (idx >= rows * 448) return;
    int i = idx / 448, c = idx % 448;
    float v = 0.0f;
    if (c < 119) {
        v = atom[(size_t)i * 119 + c];
    } else if (c < 419) {
        int h = c - 119;
#pragma unroll
        for (int j = 0; j < 6; ++j) {
            int p = idxt[i * 6 + j];
            if (p != 0) v += src[(size_t)p * 300 + h];
        }
    }
    out[idx] = (__bf16)v;
}

// out16[rows,1344] = bf16[ agg|ma|mb|atom_message|atom(119) | 0-pad ]
__global__ void k_cat_node(const float* __restrict__ agg, const float* __restrict__ ma,
                           const float* __restrict__ mb, const float* __restrict__ am,
                           const float* __restrict__ atom, __bf16* __restrict__ out, int rows)
{
    int idx = blockIdx.x * blockDim.x + threadIdx.x;
    if (idx >= rows * 1344) return;
    int i = idx / 1344, c = idx % 1344;
    float v = 0.0f;
    if (c < 300)       v = agg[(size_t)i * 300 + c];
    else if (c < 600)  v = ma[(size_t)i * 300 + (c - 300)];
    else if (c < 900)  v = mb[(size_t)i * 300 + (c - 600)];
    else if (c < 1200) v = am[(size_t)i * 300 + (c - 900)];
    else if (c < 1319) v = atom[(size_t)i * 119 + (c - 1200)];
    out[idx] = (__bf16)v;
}

__global__ void k_h0max(const float* __restrict__ node, float* __restrict__ h0)
{
    int idx = blockIdx.x * blockDim.x + threadIdx.x;
    if (idx >= 512 * 600) return;
    int b = idx / 600, h = idx % 600;
    float m = -3.4e38f;
    for (int t = 0; t < 48; ++t)
        m = fmaxf(m, node[(size_t)(1 + b * 48 + t) * 600 + h]);
    h0[idx] = m;
}

// msg16[rows,608] = bf16( relu(node + gru_bias) ), zero pad cols 600..607
__global__ void k_relu_bias(const float* __restrict__ node, const float* __restrict__ bias,
                            __bf16* __restrict__ out, int rows)
{
    int idx = blockIdx.x * blockDim.x + threadIdx.x;
    if (idx >= rows * 608) return;
    int i = idx / 608, c = idx % 608;
    float v = 0.0f;
    if (c < 600) v = fmaxf(node[(size_t)i * 600 + c] + bias[c], 0.0f);
    out[idx] = (__bf16)v;
}

// message16 row 0 = [msg16[0,:600] | msg16[0,:600] | 0-pad]
__global__ void k_row0(const __bf16* __restrict__ msg16, __bf16* __restrict__ message16)
{
    int idx = blockIdx.x * blockDim.x + threadIdx.x;
    if (idx >= 1216) return;
    message16[idx] = (idx < 1200) ? msg16[idx % 600] : (__bf16)0.0f;
}

// zero pad columns 1200..1215 of message16 for all rows
__global__ void k_zero_pad16(__bf16* __restrict__ message16, int rows)
{
    int idx = blockIdx.x * blockDim.x + threadIdx.x;
    if (idx >= rows * 16) return;
    int i = idx / 16, c = idx % 16;
    message16[(size_t)i * 1216 + 1200 + c] = (__bf16)0.0f;
}

__global__ void k_copy(const float* __restrict__ a, float* __restrict__ b, int n)
{
    int idx = blockIdx.x * blockDim.x + threadIdx.x;
    if (idx < n) b[idx] = a[idx];
}

// GI[b*48+t, 1800] (b_ih added), GH[b, 1800] (b_hh added), h[b,600] f32 state.
// Writes: f32 h (recurrent state), bf16 h (next GEMM input), bf16 message16.
__global__ void k_gru_gate(const float* __restrict__ GI, const float* __restrict__ GH,
                           float* __restrict__ h, __bf16* __restrict__ h16,
                           __bf16* __restrict__ message16, int t, int colOff)
{
    int idx = blockIdx.x * blockDim.x + threadIdx.x;
    if (idx >= 512 * 600) return;
    int b = idx / 600, hh = idx % 600;
    const float* gi = GI + (size_t)(b * 48 + t) * 1800;
    const float* gh = GH + (size_t)b * 1800;
    float ir = gi[hh], iz = gi[600 + hh], in_ = gi[1200 + hh];
    float hr = gh[hh], hz = gh[600 + hh], hn  = gh[1200 + hh];
    float r = 1.0f / (1.0f + __expf(-(ir + hr)));
    float z = 1.0f / (1.0f + __expf(-(iz + hz)));
    float n = tanhf(in_ + r * hn);
    float h2 = (1.0f - z) * n + z * h[idx];
    h[idx] = h2;
    h16[(size_t)b * 608 + hh] = (__bf16)h2;
    message16[(size_t)(1 + b * 48 + t) * 1216 + colOff + hh] = (__bf16)h2;
}

// ---------------------------------------------------------------------------
// Host orchestration
// ---------------------------------------------------------------------------
extern "C" void kernel_launch(void* const* d_in, const int* in_sizes, int n_in,
                              void* d_out, int out_size, void* d_ws, size_t ws_size,
                              hipStream_t stream)
{
    (void)in_sizes; (void)n_in; (void)out_size; (void)ws_size;

    const int NATOM = 24577, NBOND = 49153;

    const float* atom    = (const float*)d_in[0];
    const float* bond    = (const float*)d_in[1];
    const int*   a2a     = (const int*)d_in[3];
    const int*   a2b     = (const int*)d_in[4];
    const int*   b2a     = (const int*)d_in[7];
    const int*   b2revb  = (const int*)d_in[8];
    const float* tmut    = (const float*)d_in[13];
    const float* tmeth   = (const float*)d_in[14];
    const float* tge     = (const float*)d_in[15];
    const float* W_i_atom = (const float*)d_in[16];
    const float* W_i_bond = (const float*)d_in[17];
    const float* Whs[2]   = { (const float*)d_in[18], (const float*)d_in[19] };
    const float* W_nout_w = (const float*)d_in[20];
    const float* W_nout_b = (const float*)d_in[21];
    const float* W_eout_w = (const float*)d_in[22];
    const float* W_eout_b = (const float*)d_in[23];
    const float* lr_w     = (const float*)d_in[24];
    const float* gru_bias = (const float*)d_in[25];
    const float* wih_f = (const float*)d_in[26];
    const float* whh_f = (const float*)d_in[27];
    const float* bih_f = (const float*)d_in[28];
    const float* bhh_f = (const float*)d_in[29];
    const float* wih_b = (const float*)d_in[30];
    const float* whh_b = (const float*)d_in[31];
    const float* bih_b = (const float*)d_in[32];
    const float* bhh_b = (const float*)d_in[33];
    const float* W_o_w = (const float*)d_in[34];
    const float* W_o_b = (const float*)d_in[35];

    float* out = (float*)d_out;

    // ---- workspace bump allocation (bytes, 256B aligned) ----
    char* base = (char*)d_ws;
    size_t off = 0;
    auto allocB = [&](size_t bytes) {
        void* p = base + off;
        off = (off + bytes + 255) & ~(size_t)255;
        return p;
    };
    auto allocF  = [&](size_t n) { return (float*)allocB(n * 4); };
    auto allocH  = [&](size_t n) { return (__bf16*)allocB(n * 2); };

    // f32 buffers
    float* inAtom  = allocF((size_t)NATOM * 300);   // later: ma
    float* inBond  = allocF((size_t)NBOND * 300);   // message_bond (in-place)
    float* atomMsg = allocF((size_t)NATOM * 300);
    float* rev     = allocF((size_t)NBOND * 300);
    float* msgAtom = allocF((size_t)NATOM * 300);   // later: mb
    float* agg     = allocF((size_t)NATOM * 300);
    float* node    = allocF((size_t)NATOM * 600);
    float* h0      = allocF((size_t)512 * 600);
    float* hbuf    = allocF((size_t)512 * 600);
    float* GH      = allocF((size_t)512 * 1800);
    float* GI      = allocF((size_t)24576 * 1800);
    // bf16 activation buffers (rows padded to multiple-of-32 columns)
    __bf16* atom16    = allocH((size_t)NATOM * 128);
    __bf16* bond16    = allocH((size_t)NBOND * 160);
    __bf16* tmp16     = allocH((size_t)NBOND * 320);   // later: catA16
    __bf16* catB16    = allocH((size_t)NATOM * 448);
    __bf16* catNode16 = allocH((size_t)NATOM * 1344);
    __bf16* msg16     = allocH((size_t)NATOM * 608);
    __bf16* hbuf16    = allocH((size_t)512 * 608);
    __bf16* message16 = allocH((size_t)NATOM * 1216);
    __bf16* cb316     = allocH((size_t)512 * 4224);
    // bf16 weight buffers
    __bf16* wiatom16 = allocH((size_t)300 * 128);
    __bf16* wibond16 = allocH((size_t)300 * 160);
    __bf16* wh16[2]  = { allocH((size_t)300 * 320), allocH((size_t)300 * 320) };
    __bf16* wnout16  = allocH((size_t)300 * 448);
    __bf16* weout16  = allocH((size_t)300 * 448);
    __bf16* lr16     = allocH((size_t)600 * 1344);
    __bf16* wihf16   = allocH((size_t)1800 * 608);
    __bf16* whhf16   = allocH((size_t)1800 * 608);
    __bf16* wihb16   = allocH((size_t)1800 * 608);
    __bf16* whhb16   = allocH((size_t)1800 * 608);
    __bf16* wo16     = allocH((size_t)600 * 1216);
    __bf16* fcw16    = allocH((size_t)128 * 4224);
    // CNN f32 conv buffers alias GI (CNN finishes before GRU starts)
    float* cb1 = GI;
    float* cb2 = cb1 + (size_t)512 * 32 * 331;
    float* cb3 = cb2 + (size_t)512 * 64 * 108;

    auto ew = [](long long n) { return dim3((unsigned)((n + 255) / 256)); };
    auto gemm = [&](const __bf16* A, int lda, const __bf16* W, int ldw,
                    const float* bias, float* C, int ldc,
                    int M, int N, int Kp, int act) {
        dim3 g((unsigned)((N + GBN - 1) / GBN), (unsigned)((M + GBM - 1) / GBM));
        gemm_wmma_kernel<<<g, 256, 0, stream>>>(A, lda, W, ldw, bias, C, ldc, M, N, Kp, act);
    };
    auto cvt = [&](const float* in, __bf16* o16, int rows, int K, int Kp) {
        k_cvt_pad<<<ew((long long)rows * Kp), 256, 0, stream>>>(in, o16, rows, K, Kp);
    };

    // ---- weight + input conversions (bf16, zero-padded K) ----
    cvt(W_i_atom, wiatom16, 300, 119, 128);
    cvt(W_i_bond, wibond16, 300, 132, 160);
    cvt(Whs[0], wh16[0], 300, 300, 320);
    cvt(Whs[1], wh16[1], 300, 300, 320);
    cvt(W_nout_w, wnout16, 300, 419, 448);
    cvt(W_eout_w, weout16, 300, 419, 448);
    cvt(lr_w, lr16, 600, 1319, 1344);
    cvt(wih_f, wihf16, 1800, 600, 608);
    cvt(whh_f, whhf16, 1800, 600, 608);
    cvt(wih_b, wihb16, 1800, 600, 608);
    cvt(whh_b, whhb16, 1800, 600, 608);
    cvt(W_o_w, wo16, 600, 1200, 1216);
    cvt(atom, atom16, NATOM, 119, 128);
    cvt(bond, bond16, NBOND, 132, 160);

    // ===================== Phase 1: target CNNs -> cell =====================
    auto cnn = [&](const float* x, int L,
                   const float* w1, const float* b1, const float* w2, const float* b2,
                   const float* w3, const float* b3, const float* fw, const float* fb,
                   int colOff) {
        int L1 = L - 7,  P1 = L1 / 3;
        int L2 = P1 - 7, P2 = L2 / 3;
        int L3 = P2 - 7, P3 = L3 / 3;
        int flat = 128 * P3;   // 2944 / 1280 / 4224, all multiples of 32
        conv_relu_pool_kernel<<<ew(512 * 32 * P1), 256, 0, stream>>>(x,   w1, b1, cb1, 512, 1,  L,  32, P1);
        conv_relu_pool_kernel<<<ew(512 * 64 * P2), 256, 0, stream>>>(cb1, w2, b2, cb2, 512, 32, P1, 64, P2);
        conv_relu_pool_kernel<<<ew(512 * 128 * P3), 256, 0, stream>>>(cb2, w3, b3, cb3, 512, 64, P2, 128, P3);
        cvt(cb3, cb316, 512, flat, flat);
        cvt(fw, fcw16, 128, flat, flat);
        gemm(cb316, flat, fcw16, flat, fb, out + colOff, 384, 512, 128, flat, 0);
    };
    cnn(tmut,  735,  (const float*)d_in[36], (const float*)d_in[37], (const float*)d_in[38],
        (const float*)d_in[39], (const float*)d_in[40], (const float*)d_in[41],
        (const float*)d_in[42], (const float*)d_in[43], 0);
    cnn(tmeth, 377,  (const float*)d_in[44], (const float*)d_in[45], (const float*)d_in[46],
        (const float*)d_in[47], (const float*)d_in[48], (const float*)d_in[49],
        (const float*)d_in[50], (const float*)d_in[51], 128);
    cnn(tge,   1000, (const float*)d_in[52], (const float*)d_in[53], (const float*)d_in[54],
        (const float*)d_in[55], (const float*)d_in[56], (const float*)d_in[57],
        (const float*)d_in[58], (const float*)d_in[59], 256);

    // ===================== Phase 2: graph message passing ===================
    gemm(atom16, 128, wiatom16, 128, nullptr, inAtom, 300, NATOM, 300, 128, 1);
    gemm(bond16, 160, wibond16, 160, nullptr, inBond, 300, NBOND, 300, 160, 1);
    k_atom_message<<<ew((long long)NATOM * 300), 256, 0, stream>>>(inAtom, inBond, a2a, a2b, atomMsg, NATOM);
    k_rev<<<ew((long long)NBOND * 300), 256, 0, stream>>>(inBond, b2a, b2revb, rev, NBOND);

    for (int it = 0; it < 2; ++it) {
        k_agg<<<ew((long long)NATOM * 300), 256, 0, stream>>>(inBond, a2b, agg, NATOM);
        k_add<<<ew((long long)NATOM * 300), 256, 0, stream>>>(atomMsg, agg, msgAtom, NATOM * 300);
        k_tmp<<<ew((long long)NBOND * 320), 256, 0, stream>>>(msgAtom, rev, b2a, tmp16, NBOND);
        gemm(tmp16, 320, wh16[it], 320, nullptr, inBond, 300, NBOND, 300, 320, 1);
    }
    k_agg<<<ew((long long)NATOM * 300), 256, 0, stream>>>(inBond, a2b, agg, NATOM);
    k_add<<<ew((long long)NATOM * 300), 256, 0, stream>>>(msgAtom, agg, msgAtom, NATOM * 300);

    __bf16* catA16 = tmp16;  // tmp16 dead after loop
    k_gather_cat<<<ew((long long)NATOM * 448), 256, 0, stream>>>(atom, msgAtom, a2a, catA16, NATOM);
    float* ma = inAtom;      // inAtom dead
    gemm(catA16, 448, wnout16, 448, W_nout_b, ma, 300, NATOM, 300, 448, 1);
    k_gather_cat<<<ew((long long)NATOM * 448), 256, 0, stream>>>(atom, inBond, a2b, catB16, NATOM);
    float* mb = msgAtom;     // msgAtom only fed catA (done)
    gemm(catB16, 448, weout16, 448, W_eout_b, mb, 300, NATOM, 300, 448, 1);

    k_cat_node<<<ew((long long)NATOM * 1344), 256, 0, stream>>>(agg, ma, mb, atomMsg, atom, catNode16, NATOM);
    gemm(catNode16, 1344, lr16, 1344, nullptr, node, 600, NATOM, 600, 1344, 0);
    k_h0max<<<ew(512 * 600), 256, 0, stream>>>(node, h0);
    k_relu_bias<<<ew((long long)NATOM * 608), 256, 0, stream>>>(node, gru_bias, msg16, NATOM);
    k_row0<<<ew(1216), 256, 0, stream>>>(msg16, message16);
    k_zero_pad16<<<ew((long long)NATOM * 16), 256, 0, stream>>>(message16, NATOM);

    // ===================== Phase 3: bidirectional GRU =======================
    // forward
    gemm(msg16 + 608, 608, wihf16, 608, bih_f, GI, 1800, 24576, 1800, 608, 0);
    k_copy<<<ew(512 * 600), 256, 0, stream>>>(h0, hbuf, 512 * 600);
    cvt(h0, hbuf16, 512, 600, 608);
    for (int t = 0; t < 48; ++t) {
        gemm(hbuf16, 608, whhf16, 608, bhh_f, GH, 1800, 512, 1800, 608, 0);
        k_gru_gate<<<ew(512 * 600), 256, 0, stream>>>(GI, GH, hbuf, hbuf16, message16, t, 0);
    }
    // backward (reuse GI)
    gemm(msg16 + 608, 608, wihb16, 608, bih_b, GI, 1800, 24576, 1800, 608, 0);
    k_copy<<<ew(512 * 600), 256, 0, stream>>>(h0, hbuf, 512 * 600);
    cvt(h0, hbuf16, 512, 600, 608);
    for (int t = 47; t >= 0; --t) {
        gemm(hbuf16, 608, whhb16, 608, bhh_b, GH, 1800, 512, 1800, 608, 0);
        k_gru_gate<<<ew(512 * 600), 256, 0, stream>>>(GI, GH, hbuf, hbuf16, message16, t, 600);
    }

    // ===================== Output projection ================================
    gemm(message16, 1216, wo16, 1216, W_o_b, out + 512 * 384, 600, NATOM, 600, 1216, 1);
}
